// CrossModalInteractionLayer_16097537425583
// MI455X (gfx1250) — compile-verified, matrix operand order checked by reading, not run on hardware
//
#include <hip/hip_runtime.h>

// ---------------------------------------------------------------- types ----
typedef _Float16 f16;
typedef __attribute__((ext_vector_type(16))) _Float16 v16h;
typedef __attribute__((ext_vector_type(8)))  float    v8f;

union FragH  { v16h v; uint4 u[2]; f16 h[16]; };
union Frag32 { uint4 u[4]; f16 h[32]; };

#define VIS_DIM 1024
#define TXT_DIM 768
#define SH      768
#define NHEADS  12
#define HDIM    64
#define N_VIS   576
#define N_TXT   512
#define BATCH   16

__device__ __forceinline__ v8f wmma32(v16h a, v16h b, v8f c) {
  // v_wmma_f32_16x16x32_f16: (neg_a, A, neg_b, B, c_mod, C, reuse_a, reuse_b)
  return __builtin_amdgcn_wmma_f32_16x16x32_f16(false, a, false, b, (short)0, c,
                                                false, false);
}

// Load a 16x32 (rows x K) f16 fragment in CDNA5 A/B layout from a row-major
// buffer with leading dimension ld.  ISA 7.12.2: lanes 0-15 hold row=lane,
// K {0..7,16..23}; lanes 16-31 hold row=lane-16, K {8..15,24..31}.
__device__ __forceinline__ v16h load_frag(const f16* base, int ld, int row0, int k0) {
  int lane = threadIdx.x & 31;
  const f16* p = base + (size_t)(row0 + (lane & 15)) * ld + k0 + (lane >> 4) * 8;
  FragH f;
  f.u[0] = *(const uint4*)(p);
  f.u[1] = *(const uint4*)(p + 16);
  return f.v;
}

// gfx1250 async global->LDS copy of one 16-byte chunk (ASYNCcnt-tracked).
// ldsa = LDS byte offset (low 32 bits of the shared-aperture address).
__device__ __forceinline__ void async_copy_b128(unsigned ldsa, const void* gptr) {
  asm volatile("global_load_async_to_lds_b128 %0, %1, off"
               :: "v"(ldsa), "v"(gptr) : "memory");
}
__device__ __forceinline__ void async_wait0() {
  asm volatile("s_wait_asynccnt 0" ::: "memory");
}
__device__ __forceinline__ unsigned lds_addr32(const void* shared_p) {
  return (unsigned)(size_t)shared_p;   // LDS aperture: addr[31:0] = LDS offset
}

// ------------------------------------------------- weight cast+transpose ----
// W (K x N fp32, row-major) -> Wt (N x K f16, row-major) so GEMM B operand
// uses the same fragment loader as A.
__global__ __launch_bounds__(256) void castT_kernel(const float* __restrict__ W,
                                                    f16* __restrict__ Wt,
                                                    int K, int N) {
  size_t idx = (size_t)blockIdx.x * 256 + threadIdx.x;
  size_t total = (size_t)K * N;
  if (idx >= total) return;
  int k = (int)(idx / N);
  int n = (int)(idx % N);
  Wt[(size_t)n * K + k] = (f16)W[idx];
}

// ------------------------------------------------------- layernorm + cast ----
__global__ __launch_bounds__(256) void ln_cast_kernel(const float* __restrict__ x,
                                                      const float* __restrict__ g,
                                                      const float* __restrict__ b,
                                                      f16* __restrict__ y, int D) {
  __shared__ float red[2][8];
  int row = blockIdx.x;
  const float* xr = x + (size_t)row * D;
  int tid = threadIdx.x, lane = tid & 31, wid = tid >> 5;
  float s = 0.f, s2 = 0.f;
  for (int i = tid; i < D; i += 256) { float v = xr[i]; s += v; s2 += v * v; }
  #pragma unroll
  for (int m = 1; m < 32; m <<= 1) {
    s  += __shfl_xor(s,  m, 32);
    s2 += __shfl_xor(s2, m, 32);
  }
  if (lane == 0) { red[0][wid] = s; red[1][wid] = s2; }
  __syncthreads();
  float ts = 0.f, ts2 = 0.f;
  #pragma unroll
  for (int w = 0; w < 8; ++w) { ts += red[0][w]; ts2 += red[1][w]; }
  float mean = ts / (float)D;
  float var  = ts2 / (float)D - mean * mean;
  float rstd = rsqrtf(var + 1e-5f);
  f16* yr = y + (size_t)row * D;
  for (int i = tid; i < D; i += 256)
    yr[i] = (f16)((xr[i] - mean) * rstd * g[i] + b[i]);
}

// ----------------------------------------------------------- WMMA GEMM ------
// C(MxN) = A(MxK,f16,row-major) * Bt(NxK,f16,row-major)^T  [+bias]
// mode 0: write f16       mode 1: outF = residual + sigmoid(gate)*(acc+bias)
// Tiles are staged straight into LDS with gfx1250 async-to-LDS copies.
#define GBM 128
#define GBN 64
#define GBK 32
__global__ __launch_bounds__(256) void gemm_wmma_kernel(
    const f16* __restrict__ A, const f16* __restrict__ Bt,
    const float* __restrict__ bias,
    f16* __restrict__ outH, float* __restrict__ outF,
    const float* __restrict__ residual, const float* __restrict__ gate_logit,
    int M, int N, int K, int mode) {
  __shared__ f16 As[GBM * GBK];
  __shared__ f16 Bs[GBN * GBK];

  int tid = threadIdx.x, lane = tid & 31, wid = tid >> 5;
  int wm = wid >> 1, wn = wid & 1;                // 4x2 wave grid, 32x32/wave
  int mBase = blockIdx.y * GBM, nBase = blockIdx.x * GBN;

  v8f acc[2][2];
  #pragma unroll
  for (int i = 0; i < 2; ++i)
    #pragma unroll
    for (int j = 0; j < 2; ++j)
      acc[i][j] = v8f{0.f,0.f,0.f,0.f,0.f,0.f,0.f,0.f};

  for (int kb = 0; kb < K; kb += GBK) {
    // stage A tile: 128x32 f16 = 512 x 16B async chunks (2 per thread)
    #pragma unroll
    for (int it = 0; it < 2; ++it) {
      int c = tid + it * 256;
      int row = c >> 2, koff = (c & 3) * 8;
      async_copy_b128(lds_addr32(&As[row * GBK + koff]),
                      &A[(size_t)(mBase + row) * K + kb + koff]);
    }
    // stage B tile: 64x32 f16 = 256 x 16B async chunks (1 per thread)
    {
      int n = tid >> 2, koff = (tid & 3) * 8;
      async_copy_b128(lds_addr32(&Bs[n * GBK + koff]),
                      &Bt[(size_t)(nBase + n) * K + kb + koff]);
    }
    if (kb + GBK < K)  // gfx1250 global_prefetch_b8 for next A tile
      __builtin_prefetch(&A[(size_t)(mBase + (tid >> 1)) * K + kb + GBK + (tid & 1) * 16], 0, 1);
    async_wait0();     // drain this wave's ASYNCcnt before publishing
    __syncthreads();

    v16h a0 = load_frag(As, GBK, wm * 32 +  0, 0);
    v16h a1 = load_frag(As, GBK, wm * 32 + 16, 0);
    v16h b0 = load_frag(Bs, GBK, wn * 32 +  0, 0);
    v16h b1 = load_frag(Bs, GBK, wn * 32 + 16, 0);
    acc[0][0] = wmma32(a0, b0, acc[0][0]);
    acc[0][1] = wmma32(a0, b1, acc[0][1]);
    acc[1][0] = wmma32(a1, b0, acc[1][0]);
    acc[1][1] = wmma32(a1, b1, acc[1][1]);
    __syncthreads();
  }

  float gate = 1.f;
  if (mode == 1) gate = 1.f / (1.f + __expf(-gate_logit[0]));
  int half = lane >> 4, l15 = lane & 15;
  #pragma unroll
  for (int i = 0; i < 2; ++i)
    #pragma unroll
    for (int j = 0; j < 2; ++j)
      #pragma unroll
      for (int e = 0; e < 8; ++e) {
        int row = mBase + wm * 32 + i * 16 + e + half * 8;   // C layout (7.12.2)
        int col = nBase + wn * 32 + j * 16 + l15;
        float v = acc[i][j][e] + (bias ? bias[col] : 0.f);
        if (mode == 0) outH[(size_t)row * N + col] = (f16)v;
        else outF[(size_t)row * N + col] =
                 residual[(size_t)row * N + col] + gate * v;
      }
}

// ------------------------------------------------- fused flash attention ----
// grid (nq/64, NHEADS, BATCH), block 128 (4 waves, 16 q-rows per wave).
#define AKT 64
__global__ __launch_bounds__(128) void attn_kernel(
    const f16* __restrict__ Q, const f16* __restrict__ Km,
    const f16* __restrict__ V, const float* __restrict__ bias,
    const float* __restrict__ tau_p, f16* __restrict__ Out,
    int nq, int nk) {
  __shared__ f16 Ks[AKT * HDIM];        // normalized K tile, rows=key, cols=dim
  __shared__ f16 Vt[HDIM * AKT];        // V tile transposed, rows=dim, cols=key
  __shared__ f16 Ps[4 * 16 * AKT];      // per-wave softmax-P tiles

  int tid = threadIdx.x, lane = tid & 31, w = tid >> 5;
  int half = lane >> 4, l15 = lane & 15;
  int b = blockIdx.z, h = blockIdx.y;
  int qs = blockIdx.x * 64 + w * 16;    // sequence-relative first q row of wave
  float inv_tau = 1.f / log1pf(__expf(tau_p[0]));   // 1/softplus(tau_param)

  const f16* Qbase = Q  + (size_t)b * nq * SH + h * HDIM;
  const f16* Kbase = Km + (size_t)b * nk * SH + h * HDIM;
  const f16* Vbase = V  + (size_t)b * nk * SH + h * HDIM;

  // load + L2-normalize the wave's 16 q rows in register fragments
  FragH aq[2];
  aq[0].v = load_frag(Qbase, SH, qs, 0);
  aq[1].v = load_frag(Qbase, SH, qs, 32);
  float ss = 0.f;
  #pragma unroll
  for (int t = 0; t < 2; ++t)
    #pragma unroll
    for (int e = 0; e < 16; ++e) { float x = (float)aq[t].h[e]; ss += x * x; }
  ss += __shfl_xor(ss, 16, 32);         // lane l + lane l+16 hold same row
  float qsc = 1.f / fmaxf(sqrtf(ss), 1e-8f);
  #pragma unroll
  for (int t = 0; t < 2; ++t)
    #pragma unroll
    for (int e = 0; e < 16; ++e)
      aq[t].h[e] = (f16)((float)aq[t].h[e] * qsc);

  float rowmax[8], rowsum[8];
  #pragma unroll
  for (int i = 0; i < 8; ++i) { rowmax[i] = -1e30f; rowsum[i] = 0.f; }
  v8f o[4];
  #pragma unroll
  for (int j = 0; j < 4; ++j) o[j] = v8f{0.f,0.f,0.f,0.f,0.f,0.f,0.f,0.f};

  for (int k0 = 0; k0 < nk; k0 += AKT) {
    // --- stage K (L2-normalized) and V^T tiles: 2 threads per key row ---
    {
      int kr = tid >> 1, part = tid & 1;
      const f16* kp = Kbase + (size_t)(k0 + kr) * SH + part * 32;
      Frag32 kd;
      #pragma unroll
      for (int u = 0; u < 4; ++u) kd.u[u] = *(const uint4*)(kp + u * 8);
      float s2 = 0.f;
      #pragma unroll
      for (int e = 0; e < 32; ++e) { float x = (float)kd.h[e]; s2 += x * x; }
      s2 += __shfl_xor(s2, 1, 32);      // partner thread has other half of row
      float ksc = 1.f / fmaxf(sqrtf(s2), 1e-8f);
      #pragma unroll
      for (int e = 0; e < 32; ++e) kd.h[e] = (f16)((float)kd.h[e] * ksc);
      #pragma unroll
      for (int u = 0; u < 4; ++u)
        *(uint4*)(&Ks[kr * HDIM + part * 32 + u * 8]) = kd.u[u];

      const f16* vp = Vbase + (size_t)(k0 + kr) * SH + part * 32;
      Frag32 vd;
      #pragma unroll
      for (int u = 0; u < 4; ++u) vd.u[u] = *(const uint4*)(vp + u * 8);
      #pragma unroll
      for (int e = 0; e < 32; ++e)
        Vt[(part * 32 + e) * AKT + kr] = vd.h[e];
    }
    __syncthreads();

    // --- scores: 16 q-rows x 64 keys via WMMA (Ks acts as NxK operand) ---
    v8f s[4];
    #pragma unroll
    for (int j = 0; j < 4; ++j) {
      s[j] = v8f{0.f,0.f,0.f,0.f,0.f,0.f,0.f,0.f};
      v16h bk0 = load_frag(Ks, HDIM, 16 * j, 0);
      v16h bk1 = load_frag(Ks, HDIM, 16 * j, 32);
      s[j] = wmma32(aq[0].v, bk0, s[j]);
      s[j] = wmma32(aq[1].v, bk1, s[j]);
    }
    // scale by 1/tau and add bias (per-element in C layout)
    #pragma unroll
    for (int j = 0; j < 4; ++j)
      #pragma unroll
      for (int e = 0; e < 8; ++e) {
        int qrow = qs + e + half * 8;
        int kcol = k0 + 16 * j + l15;
        s[j][e] = s[j][e] * inv_tau + bias[(size_t)qrow * nk + kcol];
      }
    // --- online softmax: rows live in 16-lane halves; butterfly reduce ---
    #pragma unroll
    for (int i = 0; i < 8; ++i) {
      float tm = fmaxf(fmaxf(s[0][i], s[1][i]), fmaxf(s[2][i], s[3][i]));
      #pragma unroll
      for (int m = 1; m < 16; m <<= 1) tm = fmaxf(tm, __shfl_xor(tm, m, 32));
      float nm = fmaxf(rowmax[i], tm);
      float corr = __expf(rowmax[i] - nm);
      float ps = 0.f;
      #pragma unroll
      for (int j = 0; j < 4; ++j) {
        float p = __expf(s[j][i] - nm);
        s[j][i] = p;
        ps += p;
      }
      #pragma unroll
      for (int m = 1; m < 16; m <<= 1) ps += __shfl_xor(ps, m, 32);
      rowsum[i] = rowsum[i] * corr + ps;
      rowmax[i] = nm;
      #pragma unroll
      for (int j = 0; j < 4; ++j) o[j][i] *= corr;
    }
    // --- P through LDS into A-fragment layout (same-wave DS is in-order) ---
    f16* Pw = &Ps[w * 16 * AKT];
    #pragma unroll
    for (int j = 0; j < 4; ++j)
      #pragma unroll
      for (int e = 0; e < 8; ++e)
        Pw[(e + half * 8) * AKT + 16 * j + l15] = (f16)s[j][e];

    v16h ap0 = load_frag(Pw, AKT, 0, 0);
    v16h ap1 = load_frag(Pw, AKT, 0, 32);
    #pragma unroll
    for (int j = 0; j < 4; ++j) {
      v16h bv0 = load_frag(Vt, AKT, 16 * j, 0);
      v16h bv1 = load_frag(Vt, AKT, 16 * j, 32);
      o[j] = wmma32(ap0, bv0, o[j]);
      o[j] = wmma32(ap1, bv1, o[j]);
    }
    __syncthreads();
  }

  f16* Ob = Out + (size_t)b * nq * SH + h * HDIM;
  #pragma unroll
  for (int j = 0; j < 4; ++j)
    #pragma unroll
    for (int e = 0; e < 8; ++e) {
      int r = qs + e + half * 8;
      Ob[(size_t)r * SH + 16 * j + l15] = (f16)(o[j][e] / rowsum[e]);
    }
}

// ----------------------------------------------------------- host driver ----
extern "C" void kernel_launch(void* const* d_in, const int* in_sizes, int n_in,
                              void* d_out, int out_size, void* d_ws, size_t ws_size,
                              hipStream_t stream) {
  (void)in_sizes; (void)n_in; (void)out_size; (void)ws_size;
  const int Mv = BATCH * N_VIS;   // 9216
  const int Mt = BATCH * N_TXT;   // 8192

  const float* vision = (const float*)d_in[0];
  const float* text   = (const float*)d_in[1];
  float* outVis = (float*)d_out;
  float* outTxt = outVis + (size_t)Mv * VIS_DIM;

  char* wsb = (char*)d_ws;
  size_t off = 0;
  auto alloc = [&](size_t elems) -> f16* {
    f16* p = (f16*)(wsb + off);
    off += ((elems * sizeof(f16)) + 255) & ~(size_t)255;
    return p;
  };

  f16* lnV = alloc((size_t)Mv * VIS_DIM);
  f16* lnT = alloc((size_t)Mt * TXT_DIM);

  struct WDesc { int idx, K, N; };
  const WDesc wl[12] = {
      {6, VIS_DIM, SH}, {8, TXT_DIM, SH},               // W_v2s, W_t2s
      {10, SH, SH}, {12, SH, SH}, {14, SH, SH},          // Wq_t, Wk_vt, Wv_vt
      {16, SH, SH},                                      // Wo_t
      {18, SH, SH}, {20, SH, SH}, {22, SH, SH},          // Wq_v, Wk_tv, Wv_tv
      {24, SH, SH},                                      // Wo_v
      {30, SH, TXT_DIM}, {32, SH, VIS_DIM}};             // W_s2t, W_s2v
  f16* wT[12];
  for (int i = 0; i < 12; ++i) wT[i] = alloc((size_t)wl[i].K * wl[i].N);

  f16* Zv  = alloc((size_t)Mv * SH);
  f16* Zt  = alloc((size_t)Mt * SH);
  f16* Qt  = alloc((size_t)Mt * SH);
  f16* Kvt = alloc((size_t)Mv * SH);
  f16* Vvt = alloc((size_t)Mv * SH);
  f16* Qv  = alloc((size_t)Mv * SH);
  f16* Ktv = alloc((size_t)Mt * SH);
  f16* Vtv = alloc((size_t)Mt * SH);
  f16* At  = alloc((size_t)Mt * SH);
  f16* Av  = alloc((size_t)Mv * SH);
  f16* AtO = alloc((size_t)Mt * SH);
  f16* AvO = alloc((size_t)Mv * SH);

  // 1) weights -> f16 transposed
  for (int i = 0; i < 12; ++i) {
    size_t tot = (size_t)wl[i].K * wl[i].N;
    castT_kernel<<<dim3((unsigned)((tot + 255) / 256)), 256, 0, stream>>>(
        (const float*)d_in[wl[i].idx], wT[i], wl[i].K, wl[i].N);
  }
  // 2) layernorm + cast
  ln_cast_kernel<<<dim3(Mv), 256, 0, stream>>>(vision, (const float*)d_in[2],
                                               (const float*)d_in[3], lnV, VIS_DIM);
  ln_cast_kernel<<<dim3(Mt), 256, 0, stream>>>(text, (const float*)d_in[4],
                                               (const float*)d_in[5], lnT, TXT_DIM);

  auto gemm = [&](const f16* A, const f16* Bt, const float* bias, f16* oh,
                  float* of, const float* res, const float* gate,
                  int M, int N, int K, int mode) {
    gemm_wmma_kernel<<<dim3(N / GBN, M / GBM), 256, 0, stream>>>(
        A, Bt, bias, oh, of, res, gate, M, N, K, mode);
  };

  // 3) shared-space projections
  gemm(lnV, wT[0], (const float*)d_in[7], Zv, nullptr, nullptr, nullptr, Mv, SH, VIS_DIM, 0);
  gemm(lnT, wT[1], (const float*)d_in[9], Zt, nullptr, nullptr, nullptr, Mt, SH, TXT_DIM, 0);
  // 4) Q/K/V projections
  gemm(Zt, wT[2], (const float*)d_in[11], Qt,  nullptr, nullptr, nullptr, Mt, SH, SH, 0);
  gemm(Zv, wT[3], (const float*)d_in[13], Kvt, nullptr, nullptr, nullptr, Mv, SH, SH, 0);
  gemm(Zv, wT[4], (const float*)d_in[15], Vvt, nullptr, nullptr, nullptr, Mv, SH, SH, 0);
  gemm(Zv, wT[6], (const float*)d_in[19], Qv,  nullptr, nullptr, nullptr, Mv, SH, SH, 0);
  gemm(Zt, wT[7], (const float*)d_in[21], Ktv, nullptr, nullptr, nullptr, Mt, SH, SH, 0);
  gemm(Zt, wT[8], (const float*)d_in[23], Vtv, nullptr, nullptr, nullptr, Mt, SH, SH, 0);
  // 5) fused flash cross-attention
  attn_kernel<<<dim3(N_TXT / 64, NHEADS, BATCH), 128, 0, stream>>>(
      Qt, Kvt, Vvt, (const float*)d_in[28], (const float*)d_in[26], At, N_TXT, N_VIS);
  attn_kernel<<<dim3(N_VIS / 64, NHEADS, BATCH), 128, 0, stream>>>(
      Qv, Ktv, Vtv, (const float*)d_in[29], (const float*)d_in[27], Av, N_VIS, N_TXT);
  // 6) output projections
  gemm(At, wT[5], (const float*)d_in[17], AtO, nullptr, nullptr, nullptr, Mt, SH, SH, 0);
  gemm(Av, wT[9], (const float*)d_in[25], AvO, nullptr, nullptr, nullptr, Mv, SH, SH, 0);
  // 7) back-projection + gated residual (fp32 output)
  gemm(AtO, wT[10], (const float*)d_in[31], nullptr, outTxt, text,
       (const float*)d_in[34], Mt, TXT_DIM, SH, 1);
  gemm(AvO, wT[11], (const float*)d_in[33], nullptr, outVis, vision,
       (const float*)d_in[35], Mv, VIS_DIM, SH, 1);
}
